// GPT_17901423690555
// MI455X (gfx1250) — matrix operand order repaired and checked by reading, not compiled, not run on hardware
//
#include <hip/hip_runtime.h>
#include <hip/hip_bf16.h>

// ---------------------------------------------------------------------------
// GPT block forward for MI455X (gfx1250), wave32 + WMMA f16->f32.
// B=2, S=1024, D=1024, H=16, HD=64, F*D=4096, V=32000, L=6 (shared weights).
// ---------------------------------------------------------------------------

typedef _Float16 v16h __attribute__((ext_vector_type(16)));
typedef _Float16 v8h  __attribute__((ext_vector_type(8)));
typedef float    v8f  __attribute__((ext_vector_type(8)));

#define GPT_B  2
#define GPT_S  1024
#define GPT_D  1024
#define GPT_H  16
#define GPT_HD 64
#define GPT_V  32000
#define GPT_L  6
#define GPT_M  (GPT_B * GPT_S)   // 2048 rows

__device__ __forceinline__ v8f wmma_f16(v16h a, v16h b, v8f c) {
  return __builtin_amdgcn_wmma_f32_16x16x32_f16(false, a, false, b,
                                                (short)0, c, false, false);
}

// Build a v16h A/B fragment from two contiguous 8-half runs in LDS.
__device__ __forceinline__ v16h ld_frag2(const _Float16* p0, const _Float16* p1) {
  v8h lo = *(const v8h*)p0;
  v8h hi = *(const v8h*)p1;
  v16h r;
#pragma unroll
  for (int i = 0; i < 8; ++i) { r[i] = lo[i]; r[8 + i] = hi[i]; }
  return r;
}

// ---------------------------------------------------------------------------
// 0) One-time (per launch) weight convert+transpose: Wt[N][K] f16 <- W[K][N] f32
// ---------------------------------------------------------------------------
__global__ __launch_bounds__(256) void convt_kernel(
    const float* __restrict__ W, _Float16* __restrict__ Wt, int K, int N) {
  __shared__ float tile[32][33];
  const int tn = blockIdx.x * 32;
  const int tk = blockIdx.y * 32;
  const int lx = threadIdx.x & 31;
  const int ly = threadIdx.x >> 5;          // 0..7
#pragma unroll
  for (int i = 0; i < 32; i += 8)
    tile[ly + i][lx] = W[(long)(tk + ly + i) * N + tn + lx];
  __syncthreads();
#pragma unroll
  for (int i = 0; i < 32; i += 8)
    Wt[(long)(tn + ly + i) * K + tk + lx] = (_Float16)tile[lx][ly + i];
}

// ---------------------------------------------------------------------------
// 1) Embedding: x[b*S+s, :] = tok_emb[X[b,s], :] + pos_emb[0, s, :]
// ---------------------------------------------------------------------------
__global__ __launch_bounds__(256) void embed_kernel(
    const int* __restrict__ X, const float* __restrict__ tok,
    const float* __restrict__ pos, float* __restrict__ x) {
  const int bs = blockIdx.x;            // 0..2047
  const int s  = bs & (GPT_S - 1);
  const long trow = (long)X[bs] * GPT_D;
  const long orow = (long)bs * GPT_D;
  for (int d = threadIdx.x; d < GPT_D; d += 256)
    x[orow + d] = tok[trow + d] + pos[(long)s * GPT_D + d];
}

// ---------------------------------------------------------------------------
// 2) WMMA GEMM:  C[M,N] = act(A[M,K] @ W + bias[N])
//    Block tile 128x64, 256 threads = 8 waves, each wave does a 32x32 tile
//    (four v_wmma_f32_16x16x32_f16 per K-step of 32).
//    F16W=true : W given as pre-transposed f16  Wt[N][K]
//    F16W=false: W given as row-major f32       W[K][N]
//    ACT: 0 = none, 1 = exact GELU.
// ---------------------------------------------------------------------------
template <int ACT, bool F16W>
__global__ __launch_bounds__(256) void gemm_kernel(
    const float* __restrict__ A, const void* __restrict__ Wp,
    const float* __restrict__ bias, float* __restrict__ C,
    int M, int N, int K) {
  __shared__ __align__(16) _Float16 As[128][32];   // [row][k]
  __shared__ __align__(16) _Float16 BsT[64][32];   // [n][k]

  const int t    = threadIdx.x;
  const int lane = t & 31;
  const int w    = t >> 5;
  const int wm   = w >> 1;          // 0..3 : 32-row group
  const int wn   = w & 1;           // 0..1 : 32-col group
  const int half = lane >> 4;
  const int l16  = lane & 15;
  const long m0  = (long)blockIdx.y * 128;
  const long n0  = (long)blockIdx.x * 64;

  const _Float16* Wt = (const _Float16*)Wp;   // [N][K] when F16W
  const float*    Wf = (const float*)Wp;      // [K][N] when !F16W

  v8f acc[2][2] = {};

  for (int kt = 0; kt < K; kt += 32) {
    __syncthreads();
#pragma unroll
    for (int i = 0; i < 16; ++i) {                 // stage A tile 128x32
      int e = t + i * 256;
      int r = e >> 5, kk = e & 31;
      As[r][kk] = (_Float16)A[(m0 + r) * (long)K + kt + kk];
    }
    if constexpr (F16W) {                          // stage B tile: f16 16B copies
      int n = t >> 2, seg = t & 3;
      *(v8h*)&BsT[n][seg * 8] =
          *(const v8h*)&Wt[(n0 + n) * (long)K + kt + seg * 8];
    } else {                                       // stage B tile: f32, transpose
#pragma unroll
      for (int i = 0; i < 8; ++i) {
        int e = t + i * 256;
        int kk = e >> 6, n = e & 63;
        BsT[n][kk] = (_Float16)Wf[(long)(kt + kk) * N + n0 + n];
      }
    }
    if (kt + 32 < K) {                             // prefetch next K-tile
      __builtin_prefetch(&A[(m0 + (t >> 1)) * (long)K + kt + 32], 0, 0);
      if constexpr (F16W)
        __builtin_prefetch(&Wt[(n0 + (t & 63)) * (long)K + kt + 32], 0, 0);
      else
        __builtin_prefetch(&Wf[(long)(kt + 32 + (t & 31)) * N + n0], 0, 0);
    }
    __syncthreads();

    // A fragments: element e -> k = 16*(e>>3) + 8*half + (e&7)
    const _Float16* ar0 = &As[wm * 32 + l16][0];
    const _Float16* ar1 = &As[wm * 32 + 16 + l16][0];
    v16h a0 = ld_frag2(ar0 + 8 * half, ar0 + 16 + 8 * half);
    v16h a1 = ld_frag2(ar1 + 8 * half, ar1 + 16 + 8 * half);
    // B fragments: element e -> k = 16*(e>>3) + 8*half + (e&7)
    const _Float16* br0 = &BsT[wn * 32 + l16][8 * half];
    const _Float16* br1 = &BsT[wn * 32 + 16 + l16][8 * half];
    v16h b0 = ld_frag2(br0, br0 + 16);
    v16h b1 = ld_frag2(br1, br1 + 16);

    acc[0][0] = wmma_f16(a0, b0, acc[0][0]);
    acc[0][1] = wmma_f16(a0, b1, acc[0][1]);
    acc[1][0] = wmma_f16(a1, b0, acc[1][0]);
    acc[1][1] = wmma_f16(a1, b1, acc[1][1]);
  }

  // Epilogue: bias + activation + store. acc[s][nc][i] -> row i+8*half, col l16.
#pragma unroll
  for (int sub = 0; sub < 2; ++sub) {
#pragma unroll
    for (int nc = 0; nc < 2; ++nc) {
      long n = n0 + wn * 32 + nc * 16 + l16;
      float bv = bias[n];
#pragma unroll
      for (int i = 0; i < 8; ++i) {
        long mrow = m0 + wm * 32 + sub * 16 + 8 * half + i;
        float v = acc[sub][nc][i] + bv;
        if (ACT == 1) v = 0.5f * v * (1.0f + erff(v * 0.70710678118f));
        C[mrow * (long)N + n] = v;
      }
    }
  }
}

// ---------------------------------------------------------------------------
// 3) Flash attention (causal). qkv layout: [b, s, h*192 + {q:0,k:64,v:128}+d]
//    One block = 64 q-rows of one (b,h). 128 threads = 4 waves x 16 rows.
//    Iterates 32-key tiles: S=Q@K^T (WMMA), online softmax (base-2),
//    O += P@V (WMMA). Writes [b, s, h*64+d] into out.
// ---------------------------------------------------------------------------
__global__ __launch_bounds__(128) void flash_kernel(
    const float* __restrict__ qkv, float* __restrict__ out) {
  __shared__ __align__(16) _Float16 Ks[32][64];     // [key][d]
  __shared__ __align__(16) _Float16 Vt[64][32];     // [d][key]
  __shared__ __align__(16) _Float16 Ps[4][16][32];  // per-wave P tile

  const int t    = threadIdx.x;
  const int lane = t & 31;
  const int w    = t >> 5;
  const int half = lane >> 4;
  const int l16  = lane & 15;
  const int qb   = blockIdx.x * 64;
  const int h    = blockIdx.y;
  const int b    = blockIdx.z;
  const long base = (long)b * GPT_S * 3072;
  const int  hq   = h * 192;

  // Q fragments (HD=64 -> two K-chunks of 32), scale = 1/sqrt(64) * log2(e)
  const float qscale = 0.125f * 1.44269504f;
  const int qrow = qb + w * 16 + l16;
  const long qoff = base + (long)qrow * 3072 + hq;
  v16h aq0, aq1;
#pragma unroll
  for (int e = 0; e < 16; ++e) {
    int d0 = 16 * (e >> 3) + 8 * half + (e & 7);
    aq0[e] = (_Float16)(qkv[qoff + d0] * qscale);
    aq1[e] = (_Float16)(qkv[qoff + 32 + d0] * qscale);
  }

  float m[8], lsum[8];
  v8f o0 = {}, o1 = {}, o2 = {}, o3 = {};
#pragma unroll
  for (int i = 0; i < 8; ++i) { m[i] = -1e30f; lsum[i] = 0.0f; }

  const int jmax = (qb + 63) / 32;       // inclusive key-tile index
  for (int j = 0; j <= jmax; ++j) {
    __syncthreads();
#pragma unroll
    for (int i = 0; i < 16; ++i) {       // stage K tile + V tile (transposed)
      int e = t + i * 128;               // 2048 halfs each
      int kk = e >> 6, d = e & 63;
      long row = base + (long)(j * 32 + kk) * 3072 + hq;
      Ks[kk][d] = (_Float16)qkv[row + 64 + d];
      Vt[d][kk] = (_Float16)qkv[row + 128 + d];
    }
    __syncthreads();

    // S tile: 16 rows x 32 keys  (two 16x16 chunks, accumulate 2 K-chunks)
    v8f s0 = {}, s1 = {};
#pragma unroll
    for (int nc = 0; nc < 2; ++nc) {
      v8f accs = {};
#pragma unroll
      for (int kc = 0; kc < 2; ++kc) {
        const _Float16* p = &Ks[nc * 16 + l16][kc * 32 + 8 * half];
        v16h bb = ld_frag2(p, p + 16);
        accs = wmma_f16(kc == 0 ? aq0 : aq1, bb, accs);
      }
      if (nc == 0) s0 = accs; else s1 = accs;
    }

    // Causal mask + online softmax; stash P (f16) into per-wave LDS.
    const int qg_base = qb + w * 16;
#pragma unroll
    for (int i = 0; i < 8; ++i) {
      int row = i + 8 * half;
      int qg  = qg_base + row;
      int k0  = j * 32 + l16;
      float v0 = (k0      <= qg) ? s0[i] : -1e30f;
      float v1 = (k0 + 16 <= qg) ? s1[i] : -1e30f;
      float tmax = fmaxf(v0, v1);
#pragma unroll
      for (int off = 1; off < 16; off <<= 1)
        tmax = fmaxf(tmax, __shfl_xor(tmax, off, 16));
      float mnew = fmaxf(m[i], tmax);
      float p0 = exp2f(v0 - mnew);
      float p1 = exp2f(v1 - mnew);
      float alpha = exp2f(m[i] - mnew);
      float rs = p0 + p1;
#pragma unroll
      for (int off = 1; off < 16; off <<= 1)
        rs += __shfl_xor(rs, off, 16);
      lsum[i] = lsum[i] * alpha + rs;
      m[i] = mnew;
      o0[i] *= alpha; o1[i] *= alpha; o2[i] *= alpha; o3[i] *= alpha;
      Ps[w][row][l16]      = (_Float16)p0;
      Ps[w][row][l16 + 16] = (_Float16)p1;
    }

    // Wave-local LDS RAW: wait for the P stores before A-fragment reads.
    asm volatile("s_wait_dscnt 0x0" ::: "memory");

    const _Float16* pr = &Ps[w][l16][0];
    v16h ap = ld_frag2(pr + 8 * half, pr + 16 + 8 * half);

    // O += P @ V : four 16-col chunks over HD=64
#pragma unroll
    for (int nc = 0; nc < 4; ++nc) {
      const _Float16* p = &Vt[nc * 16 + l16][8 * half];
      v16h bb = ld_frag2(p, p + 16);
      if      (nc == 0) o0 = wmma_f16(ap, bb, o0);
      else if (nc == 1) o1 = wmma_f16(ap, bb, o1);
      else if (nc == 2) o2 = wmma_f16(ap, bb, o2);
      else              o3 = wmma_f16(ap, bb, o3);
    }
  }

  // Normalize and store: out[b, q, h*64 + d]
#pragma unroll
  for (int i = 0; i < 8; ++i) {
    float inv = 1.0f / lsum[i];
    int qg = qb + w * 16 + 8 * half + i;
    long ob = ((long)b * GPT_S + qg) * GPT_D + h * GPT_HD;
    out[ob +  0 + l16] = o0[i] * inv;
    out[ob + 16 + l16] = o1[i] * inv;
    out[ob + 32 + l16] = o2[i] * inv;
    out[ob + 48 + l16] = o3[i] * inv;
  }
}

// ---------------------------------------------------------------------------
// 4) Fused residual-add + LayerNorm over rows of D=1024 (256 threads/row).
// ---------------------------------------------------------------------------
__global__ __launch_bounds__(256) void add_ln_kernel(
    const float* __restrict__ xin, const float* __restrict__ res,
    const float* __restrict__ sc, const float* __restrict__ bi,
    float* __restrict__ xout, int hasRes) {
  __shared__ float sb1[8], sb2[8];
  const long row = blockIdx.x;
  const float* px = xin + row * GPT_D;
  float v[4], s = 0.0f, s2 = 0.0f;
#pragma unroll
  for (int i = 0; i < 4; ++i) {
    int d = threadIdx.x + i * 256;
    float a = px[d];
    if (hasRes) a += res[row * GPT_D + d];
    v[i] = a; s += a; s2 += a * a;
  }
#pragma unroll
  for (int off = 16; off; off >>= 1) { s += __shfl_xor(s, off, 32); s2 += __shfl_xor(s2, off, 32); }
  if ((threadIdx.x & 31) == 0) { sb1[threadIdx.x >> 5] = s; sb2[threadIdx.x >> 5] = s2; }
  __syncthreads();
  s = 0.0f; s2 = 0.0f;
#pragma unroll
  for (int i = 0; i < 8; ++i) { s += sb1[i]; s2 += sb2[i]; }
  const float mean = s * (1.0f / GPT_D);
  const float var  = s2 * (1.0f / GPT_D) - mean * mean;
  const float rstd = rsqrtf(var + 1e-5f);
  __syncthreads();
#pragma unroll
  for (int i = 0; i < 4; ++i) {
    int d = threadIdx.x + i * 256;
    xout[row * GPT_D + d] = (v[i] - mean) * rstd * sc[d] + bi[d];
  }
}

// ---------------------------------------------------------------------------
// 5) Row softmax over V=32000, in place.
// ---------------------------------------------------------------------------
__global__ __launch_bounds__(256) void softmax_kernel(float* __restrict__ out, int N) {
  __shared__ float sb[8];
  float* p = out + (long)blockIdx.x * N;
  const int t = threadIdx.x;

  float mx = -1e30f;
  for (int d = t; d < N; d += 256) mx = fmaxf(mx, p[d]);
#pragma unroll
  for (int off = 16; off; off >>= 1) mx = fmaxf(mx, __shfl_xor(mx, off, 32));
  if ((t & 31) == 0) sb[t >> 5] = mx;
  __syncthreads();
  mx = -1e30f;
#pragma unroll
  for (int i = 0; i < 8; ++i) mx = fmaxf(mx, sb[i]);
  __syncthreads();

  float s = 0.0f;
  for (int d = t; d < N; d += 256) { float e = expf(p[d] - mx); p[d] = e; s += e; }
#pragma unroll
  for (int off = 16; off; off >>= 1) s += __shfl_xor(s, off, 32);
  if ((t & 31) == 0) sb[t >> 5] = s;
  __syncthreads();
  s = 0.0f;
#pragma unroll
  for (int i = 0; i < 8; ++i) s += sb[i];
  const float inv = 1.0f / s;
  for (int d = t; d < N; d += 256) p[d] *= inv;
}

// ---------------------------------------------------------------------------
// Orchestration
// ---------------------------------------------------------------------------
extern "C" void kernel_launch(void* const* d_in, const int* in_sizes, int n_in,
                              void* d_out, int out_size, void* d_ws, size_t ws_size,
                              hipStream_t stream) {
  const int*   X      = (const int*)  d_in[0];
  const float* tok    = (const float*)d_in[1];
  const float* pos    = (const float*)d_in[2];
  const float* qkv_w  = (const float*)d_in[3];
  const float* qkv_b  = (const float*)d_in[4];
  const float* out_w  = (const float*)d_in[5];
  const float* out_b  = (const float*)d_in[6];
  const float* ln1_s  = (const float*)d_in[7];
  const float* ln1_b  = (const float*)d_in[8];
  const float* ln2_s  = (const float*)d_in[9];
  const float* ln2_b  = (const float*)d_in[10];
  const float* ff1_w  = (const float*)d_in[11];
  const float* ff1_b  = (const float*)d_in[12];
  const float* ff2_w  = (const float*)d_in[13];
  const float* ff2_b  = (const float*)d_in[14];
  const float* lnf_s  = (const float*)d_in[15];
  const float* lnf_b  = (const float*)d_in[16];
  const float* head_w = (const float*)d_in[17];
  const float* head_b = (const float*)d_in[18];
  float* logits = (float*)d_out;

  // Workspace layout (floats / halfs):
  //   x(2048x1024) | buf1(2048x4096) | buf2(2048x1024) | buf3(2048x1024)
  //   | f16 transposed weights: qkvT(3072x1024) outT(1024x1024)
  //                             ff1T(4096x1024) ff2T(1024x4096)
  float* x    = (float*)d_ws;
  float* buf1 = x    + (long)GPT_M * GPT_D;
  float* buf2 = buf1 + (long)GPT_M * 4096;
  float* buf3 = buf2 + (long)GPT_M * GPT_D;
  _Float16* qkvT = (_Float16*)(buf3 + (long)GPT_M * GPT_D);
  _Float16* outT = qkvT + (long)3072 * 1024;
  _Float16* ff1T = outT + (long)1024 * 1024;
  _Float16* ff2T = ff1T + (long)4096 * 1024;

  dim3 blk(256);

  // Pre-convert per-layer weights (reused 6x) to transposed f16.
  convt_kernel<<<dim3(3072 / 32, 1024 / 32), blk, 0, stream>>>(qkv_w, qkvT, 1024, 3072);
  convt_kernel<<<dim3(1024 / 32, 1024 / 32), blk, 0, stream>>>(out_w, outT, 1024, 1024);
  convt_kernel<<<dim3(4096 / 32, 1024 / 32), blk, 0, stream>>>(ff1_w, ff1T, 1024, 4096);
  convt_kernel<<<dim3(1024 / 32, 4096 / 32), blk, 0, stream>>>(ff2_w, ff2T, 4096, 1024);

  embed_kernel<<<GPT_M, blk, 0, stream>>>(X, tok, pos, x);

  for (int layer = 0; layer < GPT_L; ++layer) {
    // qkv = x @ qkv_w + qkv_b        [2048 x 3072]
    gemm_kernel<0, true><<<dim3(3072 / 64, GPT_M / 128), blk, 0, stream>>>(
        x, qkvT, qkv_b, buf1, GPT_M, 3072, GPT_D);
    // attention -> buf2              [2048 x 1024]
    flash_kernel<<<dim3(GPT_S / 64, GPT_H, GPT_B), dim3(128), 0, stream>>>(buf1, buf2);
    // proj = attn @ out_w + out_b    [2048 x 1024]
    gemm_kernel<0, true><<<dim3(GPT_D / 64, GPT_M / 128), blk, 0, stream>>>(
        buf2, outT, out_b, buf3, GPT_M, GPT_D, GPT_D);
    // x = LN(x + proj)
    add_ln_kernel<<<GPT_M, blk, 0, stream>>>(x, buf3, ln1_s, ln1_b, x, 1);
    // h1 = GELU(x @ ff1_w + ff1_b)   [2048 x 4096]
    gemm_kernel<1, true><<<dim3(4096 / 64, GPT_M / 128), blk, 0, stream>>>(
        x, ff1T, ff1_b, buf1, GPT_M, 4096, GPT_D);
    // h2 = h1 @ ff2_w + ff2_b        [2048 x 1024]
    gemm_kernel<0, true><<<dim3(GPT_D / 64, GPT_M / 128), blk, 0, stream>>>(
        buf1, ff2T, ff2_b, buf2, GPT_M, GPT_D, 4096);
    // x = LN(x + h2)
    add_ln_kernel<<<GPT_M, blk, 0, stream>>>(x, buf2, ln2_s, ln2_b, x, 1);
  }

  // xf = LN(x)
  add_ln_kernel<<<GPT_M, blk, 0, stream>>>(x, nullptr, lnf_s, lnf_b, buf3, 0);
  // logits = xf @ head_w + head_b    [2048 x 32000] (f32 weight path, used once)
  gemm_kernel<0, false><<<dim3(GPT_V / 64, GPT_M / 128), blk, 0, stream>>>(
      buf3, head_w, head_b, logits, GPT_M, GPT_V, GPT_D);
  // softmax rows in place
  softmax_kernel<<<GPT_M, blk, 0, stream>>>(logits, GPT_V);
}
